// Trittention_48567490183347
// MI455X (gfx1250) — matrix operand order repaired and checked by reading, not compiled, not run on hardware
//
#include <hip/hip_runtime.h>
#include <hip/hip_bf16.h>

#define DIMC  512
#define NHEAD 8
#define DH    64
#define TT    192
#define BSZ   2
#define HD    (NHEAD * DH)      // 512
#define PROJC (5 * HD)          // 2560
#define ROWS  (BSZ * TT)        // 384

typedef __attribute__((ext_vector_type(16))) _Float16 v16h;
typedef __attribute__((ext_vector_type(8)))  float    v8f;
typedef __attribute__((ext_vector_type(4)))  unsigned int u32x4;
typedef __attribute__((ext_vector_type(8)))  unsigned int u32x8;

__device__ __forceinline__ v8f wmma_f16(v16h a, v16h b, v8f c) {
  // (neg_a, A, neg_b, B, c_mod, C, reuse_a, reuse_b)
  return __builtin_amdgcn_wmma_f32_16x16x32_f16(false, a, false, b, (short)0, c,
                                                false, false);
}

// ---------------------------------------------------------------------------
// Tensor Data Mover: 1-D contiguous copy of `nelem` 16-bit elements from
// global memory into LDS at byte offset `lds_off`.  D# packing per
// CDNA5 ISA (count=1, type=2, data_size=2B,
// tensor_dim0 = tile_dim0 = nelem, higher dims unused).  Issued per-wave
// (EXEC ignored); completion tracked by TENSORcnt.
// ---------------------------------------------------------------------------
__device__ __forceinline__ void tdm_copy_b16(unsigned lds_off,
                                             const _Float16* gsrc,
                                             unsigned nelem) {
  const unsigned long long ga = (unsigned long long)(uintptr_t)gsrc;
  u32x4 g0;
  g0[0] = 1u;                                              // count=1 (valid)
  g0[1] = lds_off;                                         // lds_addr (bytes)
  g0[2] = (unsigned)ga;                                    // global_addr[31:0]
  g0[3] = (unsigned)((ga >> 32) & 0x01FFFFFFu)             // global_addr[56:32]
        | (2u << 30);                                      // type=2 ("image")
  u32x8 g1;
  g1[0] = 1u << 16;                                        // data_size=1 -> 2B
  g1[1] = (nelem & 0xFFFFu) << 16;                         // tensor_dim0 lo16
  g1[2] = (nelem >> 16) | (1u << 16);                      // dim0 hi16, dim1=1
  g1[3] = (nelem & 0xFFFFu) << 16;                         // tile_dim0
  g1[4] = 1u;                                              // tile_dim1=1
  g1[5] = nelem;                                           // dim0_stride lo32
  g1[6] = 0u;                                              // stride hi, dim1_stride
  g1[7] = 0u;
  asm volatile("tensor_load_to_lds %0, %1" :: "s"(g0), "s"(g1) : "memory");
}

// ---------------------------------------------------------------------------
// LayerNorm: one 256-thread block per row of 512
// ---------------------------------------------------------------------------
__global__ void ln_kernel(const float* __restrict__ x,
                          const float* __restrict__ gamma,
                          const float* __restrict__ beta,
                          float* __restrict__ xn) {
  const int row = blockIdx.x;
  const int tid = threadIdx.x;
  const float* xr = x + (size_t)row * DIMC;
  float v0 = xr[tid], v1 = xr[tid + 256];

  __shared__ float red[256];
  red[tid] = v0 + v1;
  __syncthreads();
  for (int off = 128; off > 0; off >>= 1) {
    if (tid < off) red[tid] += red[tid + off];
    __syncthreads();
  }
  const float mu = red[0] * (1.0f / DIMC);
  __syncthreads();
  const float d0 = v0 - mu, d1 = v1 - mu;
  red[tid] = d0 * d0 + d1 * d1;
  __syncthreads();
  for (int off = 128; off > 0; off >>= 1) {
    if (tid < off) red[tid] += red[tid + off];
    __syncthreads();
  }
  const float inv = rsqrtf(red[0] * (1.0f / DIMC) + 1e-5f);
  float* o = xn + (size_t)row * DIMC;
  o[tid]       = d0 * inv * gamma[tid]       + beta[tid];
  o[tid + 256] = d1 * inv * gamma[tid + 256] + beta[tid + 256];
}

// ---------------------------------------------------------------------------
// Dense GEMM: C[M,N] = A[M,K] @ Bw[N,K]^T + bias[N], f16 WMMA, fp32 accum.
// Block = 128 threads = 4 waves; each wave owns one 16x16 tile.
// Grid: (N/64, M/16). K % 32 == 0.
// Optional epilogue: segments 0/1 of the projection are also written as f16
// in [b][h][t][64] layout (auxA/auxB) so the attention kernel can stage them
// into LDS with the Tensor Data Mover (pure byte copy, no conversion).
// ---------------------------------------------------------------------------
__global__ void gemm_wmma_kernel(const float* __restrict__ A,
                                 const float* __restrict__ Bw,
                                 const float* __restrict__ bias,
                                 float* __restrict__ C, int N, int K,
                                 _Float16* __restrict__ auxA,
                                 _Float16* __restrict__ auxB) {
  const int lane = threadIdx.x & 31;
  const int wave = threadIdx.x >> 5;
  const int tn = blockIdx.x * 4 + wave;
  const int tm = blockIdx.y;
  const int r15 = lane & 15;
  const int hi  = lane >> 4;   // 0 or 1

  // A fragment rows: lane holds M = lane&15; 16 contiguous K at hi*16
  const float* arow = A  + (size_t)(tm * 16 + r15) * K + hi * 16;
  // B fragment cols: lane holds N = lane&15; 16 contiguous K at hi*16
  const float* brow = Bw + (size_t)(tn * 16 + r15) * K + hi * 16;

  v8f acc = {};
  for (int k = 0; k < K; k += 32) {
    __builtin_prefetch(arow + k + 64, 0, 0);
    __builtin_prefetch(brow + k + 64, 0, 0);
    v16h af, bf;
#pragma unroll
    for (int i = 0; i < 8; ++i) {
      af[i]     = (_Float16)arow[k + i];
      af[i + 8] = (_Float16)arow[k + 8 + i];
    }
#pragma unroll
    for (int i = 0; i < 16; ++i) bf[i] = (_Float16)brow[k + i];
    acc = wmma_f16(af, bf, acc);
  }

  const int n  = tn * 16 + r15;
  const float bv = bias[n];
#pragma unroll
  for (int v = 0; v < 8; ++v) {
    C[(size_t)(tm * 16 + v + hi * 8) * N + n] = acc[v] + bv;
  }

  if (auxA != nullptr) {
    const int seg = n >> 9;            // which of the 5 512-wide segments
    if (seg < 2) {
      _Float16* dst = seg ? auxB : auxA;
      const int hh = (n >> 6) & 7;
      const int d  = n & 63;
#pragma unroll
      for (int v = 0; v < 8; ++v) {
        const int m  = tm * 16 + v + hi * 8;
        const int bb = m / TT, l = m % TT;   // 16-row tiles never straddle b
        dst[((size_t)(bb * NHEAD + hh) * TT + l) * DH + d] =
            (_Float16)(acc[v] + bv);
      }
    }
  }
}

// ---------------------------------------------------------------------------
// Trittention core: one 256-thread (8-wave) workgroup per (b, h, q).
//   S_q[l,r] = ((A ∘ c_q) @ B^T)[l,r] / 64      (12x12 grid of 16x16 WMMA tiles)
//   softmax over the joint (l,r) plane, then marginal reductions:
//   z_q = (rowsum @ D + colsum @ E) / Z
// Full S (144KB) lives in LDS; nothing t^3-sized ever touches HBM.
// A/B f16 slabs (24KB each, contiguous per (b,h)) are staged into LDS by the
// Tensor Data Mover from wave 0, synchronized via TENSORcnt + barrier.
// ---------------------------------------------------------------------------
__global__ void trittention_kernel(const float* __restrict__ proj,
                                   const _Float16* __restrict__ Ah,
                                   const _Float16* __restrict__ Bh,
                                   float* __restrict__ z) {
  const int q = blockIdx.x;
  const int h = blockIdx.y;
  const int b = blockIdx.z;
  const int tid  = threadIdx.x;     // 256
  const int lane = tid & 31;
  const int wave = tid >> 5;        // 0..7

  extern __shared__ char smem[];
  float*    S      = (float*)smem;                 // TT*TT fp32      (147456 B)
  _Float16* Ash    = (_Float16*)(S + TT * TT);     // TT*DH f16       ( 24576 B)
  _Float16* Bsh    = Ash + TT * DH;                // TT*DH f16       ( 24576 B)
  float*    cq     = (float*)(Bsh + TT * DH);      // DH fp32
  float*    red    = cq + DH;                      // 256 fp32
  float*    rowsum = red + 256;                    // TT fp32
  float*    colsum = rowsum + TT;                  // TT fp32

  const float* pb = proj + (size_t)(b * TT) * PROJC + h * DH;
  const size_t slab = (size_t)(b * NHEAD + h) * TT * DH;

  // Wave 0 issues two TDM copies (a-slab, b-slab) into LDS and waits on
  // TENSORcnt; the workgroup barrier publishes LDS to the other waves.
  if (tid < 32) {
    tdm_copy_b16((unsigned)((char*)Ash - smem), Ah + slab, TT * DH);
    tdm_copy_b16((unsigned)((char*)Bsh - smem), Bh + slab, TT * DH);
    __builtin_amdgcn_s_wait_tensorcnt(0);
  }
  if (tid < DH) cq[tid] = pb[(size_t)q * PROJC + 2 * HD + tid];
  __syncthreads();

  // ---- Pass 1: score tiles via WMMA (a ∘ c_q fused into A-fragment build)
  const int r15 = lane & 15;
  const int hi  = lane >> 4;
  float tmax = -3.0e38f;
  for (int tile = wave; tile < (TT / 16) * (TT / 16); tile += 8) {
    const int lt = tile / (TT / 16);
    const int rt = tile % (TT / 16);
    const _Float16* ar = Ash + (size_t)(lt * 16 + r15) * DH;
    const _Float16* br = Bsh + (size_t)(rt * 16 + r15) * DH;
    v8f acc = {};
#pragma unroll
    for (int kc = 0; kc < DH; kc += 32) {
      v16h af;
      const int ao = kc + hi * 16;
#pragma unroll
      for (int i = 0; i < 8; ++i) {
        af[i]     = (_Float16)((float)ar[ao + i]      * cq[ao + i]);
        af[i + 8] = (_Float16)((float)ar[ao + 8 + i] * cq[ao + 8 + i]);
      }
      const v16h bf = *(const v16h*)(br + kc + hi * 16);   // 32B aligned
      acc = wmma_f16(af, bf, acc);
    }
#pragma unroll
    for (int v = 0; v < 8; ++v) {
      const float s = acc[v] * (1.0f / DH);
      S[(size_t)(lt * 16 + v + hi * 8) * TT + rt * 16 + r15] = s;
      tmax = fmaxf(tmax, s);
    }
  }
  red[tid] = tmax;
  __syncthreads();
  for (int off = 128; off > 0; off >>= 1) {
    if (tid < off) red[tid] = fmaxf(red[tid], red[tid + off]);
    __syncthreads();
  }
  const float m = red[0];
  __syncthreads();

  // ---- Pass 2: exp in-place (float4 rows) + row sums, then column sums + Z
  if (tid < TT) {
    float rs = 0.f;
    float4* Sr = (float4*)(S + (size_t)tid * TT);
    for (int r = 0; r < TT / 4; ++r) {
      float4 v = Sr[r];
      v.x = __expf(v.x - m);
      v.y = __expf(v.y - m);
      v.z = __expf(v.z - m);
      v.w = __expf(v.w - m);
      Sr[r] = v;
      rs += (v.x + v.y) + (v.z + v.w);
    }
    rowsum[tid] = rs;
  }
  __syncthreads();
  if (tid < TT) {
    float cs = 0.f;
    for (int l = 0; l < TT; ++l) cs += S[(size_t)l * TT + tid];
    colsum[tid] = cs;
  }
  red[tid] = (tid < TT) ? rowsum[tid] : 0.f;
  __syncthreads();
  for (int off = 128; off > 0; off >>= 1) {
    if (tid < off) red[tid] += red[tid + off];
    __syncthreads();
  }
  const float invZ = 1.0f / red[0];
  __syncthreads();

  // ---- Pass 3: z[d] = invZ * Σ_l (rowsum[l]·D[l,d] + colsum[l]·E[l,d])
  const int d     = tid & 63;
  const int chunk = tid >> 6;          // 4 chunks of 48 rows
  float acc = 0.f;
  for (int l = chunk * 48; l < (chunk + 1) * 48; ++l) {
    const float dv = pb[(size_t)l * PROJC + 3 * HD + d];
    const float ev = pb[(size_t)l * PROJC + 4 * HD + d];
    acc += rowsum[l] * dv + colsum[l] * ev;
  }
  red[tid] = acc;
  __syncthreads();
  if (chunk == 0) {
    const float tot = red[d] + red[64 + d] + red[128 + d] + red[192 + d];
    z[(size_t)(b * TT + q) * HD + h * DH + d] = tot * invZ;
  }
}

// ---------------------------------------------------------------------------
extern "C" void kernel_launch(void* const* d_in, const int* in_sizes, int n_in,
                              void* d_out, int out_size, void* d_ws, size_t ws_size,
                              hipStream_t stream) {
  (void)in_sizes; (void)n_in; (void)out_size; (void)ws_size;
  const float* x       = (const float*)d_in[0];
  const float* gamma   = (const float*)d_in[1];
  const float* beta    = (const float*)d_in[2];
  const float* W_abcde = (const float*)d_in[3];
  const float* b_abcde = (const float*)d_in[4];
  const float* W_out   = (const float*)d_in[5];
  const float* b_out   = (const float*)d_in[6];
  float* out = (float*)d_out;

  float*    xn   = (float*)d_ws;                    // ROWS*DIMC fp32
  float*    proj = xn + (size_t)ROWS * DIMC;        // ROWS*PROJC fp32
  float*    zbuf = proj + (size_t)ROWS * PROJC;     // ROWS*HD fp32
  _Float16* Ah   = (_Float16*)(zbuf + (size_t)ROWS * HD);   // BSZ*NHEAD*TT*DH
  _Float16* Bh   = Ah + (size_t)BSZ * NHEAD * TT * DH;

  // 1) LayerNorm
  ln_kernel<<<ROWS, 256, 0, stream>>>(x, gamma, beta, xn);

  // 2) Joint projection: [384,512] @ [2560,512]^T + bias, with f16 sidecar
  //    for segments a/b in TDM-friendly [b][h][t][64] layout.
  gemm_wmma_kernel<<<dim3(PROJC / 64, ROWS / 16), 128, 0, stream>>>(
      xn, W_abcde, b_abcde, proj, PROJC, DIMC, Ah, Bh);

  // 3) Trilinear attention, one workgroup per (b,h,q); ~195KB LDS of 320KB WGP
  const size_t smem_bytes = (size_t)TT * TT * 4      // S
                          + (size_t)2 * TT * DH * 2  // Ash, Bsh (f16)
                          + DH * 4                   // cq
                          + 256 * 4                  // red
                          + (size_t)2 * TT * 4;      // rowsum, colsum
  trittention_kernel<<<dim3(TT, NHEAD, BSZ), 256, smem_bytes, stream>>>(
      proj, Ah, Bh, zbuf);

  // 4) Output projection: [384,512] @ [512,512]^T + bias
  gemm_wmma_kernel<<<dim3(DIMC / 64, ROWS / 16), 128, 0, stream>>>(
      zbuf, W_out, b_out, out, DIMC, HD, nullptr, nullptr);
}
